// MultiHeadAttentionClassical_65481071400481
// MI455X (gfx1250) — matrix operand order repaired and checked by reading, not compiled, not run on hardware
//
#include <hip/hip_runtime.h>
#include <hip/hip_bf16.h>
#include <math.h>

typedef __attribute__((ext_vector_type(16))) __bf16 v16bf;
typedef __attribute__((ext_vector_type(8)))  float  v8f;
typedef __attribute__((ext_vector_type(4)))  unsigned int u32x4;

union Frag16 { v16bf v; u32x4 q[2]; };   // 16 bf16 = 32B = two b128 chunks
union Pack8  { u32x4 q; __bf16 e[8]; };

#define WMMA_BF16(A, B, C) \
  __builtin_amdgcn_wmma_f32_16x16x32_bf16(false, (A), false, (B), (short)0, (C), false, false)

static constexpr int E  = 1024;   // embed dim (= N = K of projections)
static constexpr int S  = 2048;   // seq len
static constexpr int NB = 4;      // batch
static constexpr int H  = 16;     // heads
static constexpr int DK = 64;     // head dim

// CDNA5 async global->LDS copy (ASYNCcnt path). lds_off = LDS byte offset
// (low 32 bits of the generic shared address), gptr = 64-bit global address.
__device__ __forceinline__ void async_ld_b128(unsigned lds_off, const void* gptr) {
  asm volatile("global_load_async_to_lds_b128 %0, %1, off"
               :: "v"(lds_off), "v"((unsigned long long)(uintptr_t)gptr)
               : "memory");
}
__device__ __forceinline__ void wait_async0() {
  asm volatile("s_wait_asynccnt 0x0" ::: "memory");
}

// ---------------------------------------------------------------------------
// fp32 -> bf16 conversion
// ---------------------------------------------------------------------------
__global__ void cvt_f32_to_bf16(const float* __restrict__ src,
                                __bf16* __restrict__ dst, int n) {
  int i = blockIdx.x * blockDim.x + threadIdx.x;
  if (i < n) dst[i] = (__bf16)src[i];
}

// ---------------------------------------------------------------------------
// C[M,1024] = A[M,1024] * W[1024,1024]^T   (y = x @ W.T), bf16 in, bf16/f32 out
// Block: 256 threads (8 waves). Block tile 128x128, K-step 32.
// Double-buffered LDS fed by global_load_async_to_lds_b128; one barrier/step.
// Row stride padded to 40 elems (80B) to spread rows across LDS banks.
// ---------------------------------------------------------------------------
template <bool OUT_F32>
__global__ __launch_bounds__(256) void gemm_xwt(const __bf16* __restrict__ A,
                                                const __bf16* __restrict__ W,
                                                void* __restrict__ Cout, int M) {
  constexpr int LDA = 40;                 // padded row stride (elems)
  __shared__ __bf16 lA[2][128 * LDA];
  __shared__ __bf16 lB[2][128 * LDA];

  const int tid  = threadIdx.x;
  const int lane = tid & 31;
  const int w    = tid >> 5;
  const int wy   = w >> 1;        // 0..3
  const int wx   = w & 1;         // 0..1
  const int l15  = lane & 15;
  const int half = lane >> 4;

  const int bm = blockIdx.y * 128;
  const int bn = blockIdx.x * 128;

  const int lr = tid >> 1;        // 0..127 : tile row this thread stages
  const int lh = tid & 1;         // 0..1   : which 16-elem half of the 32-K row

  v8f acc[2][4];
#pragma unroll
  for (int mi = 0; mi < 2; ++mi)
#pragma unroll
    for (int nj = 0; nj < 4; ++nj)
#pragma unroll
      for (int i = 0; i < 8; ++i) acc[mi][nj][i] = 0.0f;

  auto stage = [&](int buf, int kt) {
    const __bf16* ga = A + (size_t)(bm + lr) * E + kt + lh * 16;
    const __bf16* gw = W + (size_t)(bn + lr) * E + kt + lh * 16;
    const unsigned sa = (unsigned)(uintptr_t)&lA[buf][lr * LDA + lh * 16];
    const unsigned sb = (unsigned)(uintptr_t)&lB[buf][lr * LDA + lh * 16];
    async_ld_b128(sa,      ga);
    async_ld_b128(sa + 16, ga + 8);
    async_ld_b128(sb,      gw);
    async_ld_b128(sb + 16, gw + 8);
  };

  stage(0, 0);

  constexpr int NIT = E / 32;
  for (int it = 0; it < NIT; ++it) {
    wait_async0();
    __syncthreads();
    if (it + 1 < NIT) stage((it + 1) & 1, (it + 1) * 32);
    const int buf = it & 1;

    // ---- A fragments (16x32, row = L&15, contiguous K = half*16 + e, as in B)
    Frag16 af[2];
#pragma unroll
    for (int mi = 0; mi < 2; ++mi) {
      const __bf16* base = &lA[buf][(wy * 32 + mi * 16 + l15) * LDA];
      af[mi].q[0] = *(const u32x4*)(base + half * 16);
      af[mi].q[1] = *(const u32x4*)(base + half * 16 + 8);
    }
    // ---- B fragments (32x16, col = L&15, contiguous K = half*16 + e, as in A)
    Frag16 bf[4];
#pragma unroll
    for (int nj = 0; nj < 4; ++nj) {
      const __bf16* base = &lB[buf][(wx * 64 + nj * 16 + l15) * LDA] + half * 16;
      bf[nj].q[0] = ((const u32x4*)base)[0];
      bf[nj].q[1] = ((const u32x4*)base)[1];
    }
#pragma unroll
    for (int mi = 0; mi < 2; ++mi)
#pragma unroll
      for (int nj = 0; nj < 4; ++nj)
        acc[mi][nj] = WMMA_BF16(af[mi].v, bf[nj].v, acc[mi][nj]);
  }

  // ---- epilogue: C layout row M = r + 8*half, col N = L&15
#pragma unroll
  for (int mi = 0; mi < 2; ++mi)
#pragma unroll
    for (int nj = 0; nj < 4; ++nj)
#pragma unroll
      for (int r = 0; r < 8; ++r) {
        const int row = bm + wy * 32 + mi * 16 + r + 8 * half;
        const int col = bn + wx * 64 + nj * 16 + l15;
        if (OUT_F32)
          ((float*)Cout)[(size_t)row * E + col] = acc[mi][nj][r];
        else
          ((__bf16*)Cout)[(size_t)row * E + col] = (__bf16)acc[mi][nj][r];
      }
}

// ---------------------------------------------------------------------------
// Flash attention: grid (S/128, B*H). 256 threads = 8 waves.
// Wave w owns query rows [qb + w*16, +16). Streams K/V in 128-row tiles.
// U region is reused: K tile (row stride 72) during the S phase, then the
// per-wave P stripes (row stride 136) during the PV phase.
// ---------------------------------------------------------------------------
__global__ __launch_bounds__(256) void attn_flash(const __bf16* __restrict__ Q,
                                                  const __bf16* __restrict__ Km,
                                                  const __bf16* __restrict__ Vm,
                                                  __bf16* __restrict__ O) {
  constexpr int LDK = 72;              // padded K-tile row stride (elems)
  constexpr int LDP = 136;             // padded P / Vt row stride (elems)
  __shared__ __bf16 U[8 * 16 * LDP];   // 34 KB: K tile | P stripes (time-shared)
  __shared__ __bf16 lV[64 * LDP];      // 17 KB: V^T tile [d][s]

  const int bh = blockIdx.y;
  const int b  = bh >> 4;
  const int h  = bh & 15;
  const int qb = blockIdx.x * 128;

  const int tid  = threadIdx.x;
  const int lane = tid & 31;
  const int w    = tid >> 5;
  const int l15  = lane & 15;
  const int half = lane >> 4;
  const int lr   = tid >> 1;   // 0..127
  const int lh   = tid & 1;

  const size_t headOff = (size_t)b * S * E + (size_t)h * DK;

  // ---- Q fragments for this wave's 16 rows (2 K-steps of 32 over d=64)
  Frag16 qa[2];
  {
    const __bf16* qrow = Q + headOff + (size_t)(qb + w * 16 + l15) * E;
#pragma unroll
    for (int ks = 0; ks < 2; ++ks) {
      qa[ks].q[0] = *(const u32x4*)(qrow + ks * 32 + half * 16);
      qa[ks].q[1] = *(const u32x4*)(qrow + ks * 32 + half * 16 + 8);
    }
  }

  float m[8], lsum[8];
  v8f o[4];
#pragma unroll
  for (int r = 0; r < 8; ++r) { m[r] = -INFINITY; lsum[r] = 0.0f; }
#pragma unroll
  for (int t = 0; t < 4; ++t)
#pragma unroll
    for (int i = 0; i < 8; ++i) o[t][i] = 0.0f;

  __bf16* pw = U + (size_t)w * 16 * LDP;

  for (int kb = 0; kb < S; kb += 128) {
    // ---- stage K tile [128][64] into U (row stride LDK) via async copies
    {
      const __bf16* gk = Km + headOff + (size_t)(kb + lr) * E + lh * 32;
      const unsigned sk = (unsigned)(uintptr_t)&U[lr * LDK + lh * 32];
#pragma unroll
      for (int j = 0; j < 4; ++j) async_ld_b128(sk + j * 16, gk + j * 8);
    }
    // ---- stage V tile transposed: lV[d][s]
    {
      const u32x4* gV = (const u32x4*)(Vm + headOff + (size_t)(kb + lr) * E + lh * 32);
#pragma unroll
      for (int j = 0; j < 4; ++j) {
        Pack8 p; p.q = gV[j];
#pragma unroll
        for (int i = 0; i < 8; ++i) lV[(lh * 32 + j * 8 + i) * LDP + lr] = p.e[i];
      }
    }
    wait_async0();
    __syncthreads();

    // ---- S = Q * K^T  (wave stripe 16x128 = 8 accumulator tiles)
    v8f s[8];
#pragma unroll
    for (int nj = 0; nj < 8; ++nj) {
#pragma unroll
      for (int i = 0; i < 8; ++i) s[nj][i] = 0.0f;
#pragma unroll
      for (int ks = 0; ks < 2; ++ks) {
        Frag16 kf;
        const __bf16* base = U + (size_t)(nj * 16 + l15) * LDK + ks * 32 + half * 16;
        kf.q[0] = ((const u32x4*)base)[0];
        kf.q[1] = ((const u32x4*)base)[1];
        s[nj] = WMMA_BF16(qa[ks].v, kf.v, s[nj]);
      }
      s[nj] = s[nj] * 0.125f;   // 1/sqrt(64)
    }
    __syncthreads();            // K tile dead; U becomes the P region

    // ---- online softmax (row = r + 8*half; reduce across the 16-lane half)
    float al[8];
#pragma unroll
    for (int r = 0; r < 8; ++r) {
      float mx = -INFINITY;
#pragma unroll
      for (int nj = 0; nj < 8; ++nj) mx = fmaxf(mx, s[nj][r]);
#pragma unroll
      for (int off = 1; off < 16; off <<= 1)
        mx = fmaxf(mx, __shfl_xor(mx, off, 32));
      const float mn = fmaxf(m[r], mx);
      al[r] = __expf(m[r] - mn);
      m[r]  = mn;
    }

    float psum[8];
#pragma unroll
    for (int r = 0; r < 8; ++r) psum[r] = 0.0f;
#pragma unroll
    for (int nj = 0; nj < 8; ++nj)
#pragma unroll
      for (int r = 0; r < 8; ++r) {
        const float p = __expf(s[nj][r] - m[r]);
        psum[r] += p;
        pw[(size_t)(r + 8 * half) * LDP + nj * 16 + l15] = (__bf16)p;
      }
#pragma unroll
    for (int r = 0; r < 8; ++r) {
      float ps = psum[r];
#pragma unroll
      for (int off = 1; off < 16; off <<= 1) ps += __shfl_xor(ps, off, 32);
      lsum[r] = lsum[r] * al[r] + ps;
#pragma unroll
      for (int t = 0; t < 4; ++t) o[t][r] *= al[r];
    }

    // wave-private P round-trip through LDS: wait for our ds stores
    asm volatile("s_wait_dscnt 0x0" ::: "memory");

    // ---- O += P * V   (P: 16x128 A-matrix, V^T in LDS as B-matrix)
#pragma unroll
    for (int ks = 0; ks < 4; ++ks) {
      Frag16 pf;
      const __bf16* pb = pw + (size_t)l15 * LDP + ks * 32;
      pf.q[0] = *(const u32x4*)(pb + half * 16);
      pf.q[1] = *(const u32x4*)(pb + half * 16 + 8);
#pragma unroll
      for (int t = 0; t < 4; ++t) {
        Frag16 vf;
        const __bf16* vb = lV + (size_t)(t * 16 + l15) * LDP + ks * 32 + half * 16;
        vf.q[0] = ((const u32x4*)vb)[0];
        vf.q[1] = ((const u32x4*)vb)[1];
        o[t] = WMMA_BF16(pf.v, vf.v, o[t]);
      }
    }
    __syncthreads();            // P stripes consumed; U free for next K tile
  }

  // ---- finalize: out[b, qrow, h, d] (contiguous embed layout), bf16
#pragma unroll
  for (int r = 0; r < 8; ++r) {
    const float inv = 1.0f / lsum[r];
#pragma unroll
    for (int t = 0; t < 4; ++t) {
      const int row = qb + w * 16 + r + 8 * half;
      const int col = t * 16 + l15;
      O[headOff + (size_t)row * E + col] = (__bf16)(o[t][r] * inv);
    }
  }
}

// ---------------------------------------------------------------------------
extern "C" void kernel_launch(void* const* d_in, const int* in_sizes, int n_in,
                              void* d_out, int out_size, void* d_ws, size_t ws_size,
                              hipStream_t stream) {
  const float* x  = (const float*)d_in[0];
  const float* Wq = (const float*)d_in[1];
  const float* Wk = (const float*)d_in[2];
  const float* Wv = (const float*)d_in[3];
  const float* Wo = (const float*)d_in[4];
  float* out = (float*)d_out;

  const size_t XN = (size_t)NB * S * E;   // 8388608
  const size_t WN = (size_t)E * E;        // 1048576

  __bf16* xb = (__bf16*)d_ws;
  __bf16* wq = xb + XN;
  __bf16* wk = wq + WN;
  __bf16* wv = wk + WN;
  __bf16* wo = wv + WN;
  __bf16* q  = wo + WN;
  __bf16* k  = q + XN;
  __bf16* v  = k + XN;
  __bf16* ob = v + XN;

  cvt_f32_to_bf16<<<(XN + 255) / 256, 256, 0, stream>>>(x, xb, (int)XN);
  cvt_f32_to_bf16<<<(WN + 255) / 256, 256, 0, stream>>>(Wq, wq, (int)WN);
  cvt_f32_to_bf16<<<(WN + 255) / 256, 256, 0, stream>>>(Wk, wk, (int)WN);
  cvt_f32_to_bf16<<<(WN + 255) / 256, 256, 0, stream>>>(Wv, wv, (int)WN);
  cvt_f32_to_bf16<<<(WN + 255) / 256, 256, 0, stream>>>(Wo, wo, (int)WN);

  const int M = NB * S;                    // 8192
  dim3 gg(E / 128, M / 128);               // (8, 64)
  gemm_xwt<false><<<gg, 256, 0, stream>>>(xb, wq, q, M);
  gemm_xwt<false><<<gg, 256, 0, stream>>>(xb, wk, k, M);
  gemm_xwt<false><<<gg, 256, 0, stream>>>(xb, wv, v, M);

  attn_flash<<<dim3(S / 128, NB * H), 256, 0, stream>>>(q, k, v, ob);

  gemm_xwt<true><<<gg, 256, 0, stream>>>(ob, wo, out, M);
}
